// StateNetwork_1434519076954
// MI455X (gfx1250) — matrix operand order repaired and checked by reading, not compiled, not run on hardware
//
#include <hip/hip_runtime.h>
#include <hip/hip_bf16.h>

typedef __attribute__((ext_vector_type(16))) _Float16 v16h;
typedef __attribute__((ext_vector_type(8)))  _Float16 v8h;
typedef __attribute__((ext_vector_type(8)))  float    v8f;

constexpr int BATCH = 16;
constexpr int NN    = 2048;   // tokens in net branch
constexpr int DN    = 64;     // D_NET
constexpr int DQ    = 256;    // 4*D_NET
constexpr int HIDN  = 128;
constexpr int SEQL  = 18;     // L_SFC + 2
constexpr int NROWS = NN + SEQL;  // 2066
constexpr int WAVES = 8;          // waves per flash block
constexpr int QROWS = WAVES * 16; // 128 query rows per block

__device__ __forceinline__ v8f wmma_f16(v16h a, v16h b, v8f c) {
    return __builtin_amdgcn_wmma_f32_16x16x32_f16(false, a, false, b, (short)0, c, false, false);
}

// ---------------------------------------------------------------------------
// Convert f32 weights to f16 workspace copies.
// ---------------------------------------------------------------------------
__global__ void cvt_weights(const float* Wq, const float* Wk, const float* Wv,
                            const float* Wn,
                            _Float16* wqh, _Float16* wkh, _Float16* wvh, _Float16* wnh) {
    int i = blockIdx.x * blockDim.x + threadIdx.x;
    if (i < DQ * DN) {
        wqh[i] = (_Float16)Wq[i];
        wkh[i] = (_Float16)Wk[i];
        wvh[i] = (_Float16)Wv[i];
    }
    if (i < HIDN * DQ) wnh[i] = (_Float16)Wn[i];
}

// ---------------------------------------------------------------------------
// QKV projections via WMMA. One wave (32 threads) per 16-row tile.
// Q scaled by 1/sqrt(64). V stored transposed: Vth[b][d][n].
// ---------------------------------------------------------------------------
__global__ void __launch_bounds__(32)
qkv_proj(const float* __restrict__ X,
         const _Float16* __restrict__ Wqh, const _Float16* __restrict__ Wkh,
         const _Float16* __restrict__ Wvh,
         const float* __restrict__ bq, const float* __restrict__ bk,
         const float* __restrict__ bv,
         _Float16* __restrict__ Qh, _Float16* __restrict__ Kh,
         _Float16* __restrict__ Vth) {
    const int wid = blockIdx.x;
    const int b   = wid >> 7;          // 128 row-tiles per batch
    const int q0  = (wid & 127) << 4;
    const int lane = threadIdx.x;
    const int lm   = lane & 15;
    const int ah   = (lane >> 4) * 16;  // A-operand k-half offset (halves)
    const int bh   = (lane >> 4) * 16;  // B-operand k-half offset (halves)
    const int mo   = (lane >> 4) * 8;   // C/D row offset

    // Build A tiles (16x32 f16) for the 16xDN input block, converted from f32.
    const float* xr = X + ((size_t)b * NN + q0 + lm) * DN;
    v16h a0, a1;
    for (int t = 0; t < 8; ++t) {
        a0[t]     = (_Float16)xr[0 + ah + t];
        a0[8 + t] = (_Float16)xr[8 + ah + t];
        a1[t]     = (_Float16)xr[32 + ah + t];
        a1[8 + t] = (_Float16)xr[40 + ah + t];
    }

    auto do_mat = [&](const _Float16* W, const float* bias, _Float16* Out,
                      float scale, bool transposed) {
        for (int t = 0; t < 16; ++t) {
            const int n0 = t * 16;
            const _Float16* wr = W + (size_t)(n0 + lm) * DN;  // row = output channel
            v16h b0 = *(const v16h*)(wr + bh);
            v16h b1 = *(const v16h*)(wr + 32 + bh);
            v8f acc = {0.f, 0.f, 0.f, 0.f, 0.f, 0.f, 0.f, 0.f};
            acc = wmma_f16(a0, b0, acc);
            acc = wmma_f16(a1, b1, acc);
            const float bb = bias[n0 + lm];
            for (int r = 0; r < 8; ++r) {
                const float val = (acc[r] + bb) * scale;
                const int m = r + mo;
                if (!transposed)
                    Out[((size_t)b * NN + q0 + m) * DQ + n0 + lm] = (_Float16)val;
                else
                    Out[((size_t)b * DQ + n0 + lm) * NN + q0 + m] = (_Float16)val;
            }
        }
    };
    do_mat(Wqh, bq, Qh, 0.125f, false);   // fold 1/sqrt(D_NET) into Q
    do_mat(Wkh, bk, Kh, 1.0f, false);
    do_mat(Wvh, bv, Vth, 1.0f, true);
}

// ---------------------------------------------------------------------------
// Flash attention + fused output projection.
// 8 waves / block; K,V tiles staged cooperatively into LDS (once per block,
// not once per wave) to cut L2 traffic 8x. LDS map (halves):
//   [0,8192)      K tile  : 32 keys x 256 d, row-major
//   [8192,16384)  V tile  : 256 d x 32 keys, row-major (from transposed V)
//   [16384+w*512) per-wave 16x32 P/O transpose buffer
// ---------------------------------------------------------------------------
__global__ void __launch_bounds__(WAVES * 32)
flash_kernel(const _Float16* __restrict__ Qh, const _Float16* __restrict__ Kh,
             const _Float16* __restrict__ Vth, const _Float16* __restrict__ Wnh,
             const float* __restrict__ net_b, float* __restrict__ out) {
    __shared__ __align__(16) _Float16 kv[16384 + WAVES * 512];
    constexpr int KOFF = 0;
    constexpr int VOFF = 8192;

    const int tid  = threadIdx.x;
    const int wave = tid >> 5;
    const int lane = tid & 31;
    const int blk  = blockIdx.x;
    const int b    = blk >> 4;                       // 16 blocks per batch
    const int q0   = (blk & 15) * QROWS + wave * 16; // this wave's query tile
    const int lm   = lane & 15;
    const int ah   = (lane >> 4) * 16;
    const int bh   = (lane >> 4) * 16;
    const int mo   = (lane >> 4) * 8;

    const _Float16* kb = Kh  + (size_t)b * NN * DQ;
    const _Float16* vb = Vth + (size_t)b * DQ * NN;
    _Float16* pbuf = &kv[16384 + wave * 512];

    // Preload Q as 8 A-operand chunks (16x32 each).
    v16h aq[8];
    const _Float16* qr = Qh + ((size_t)b * NN + q0 + lm) * DQ;
    for (int c = 0; c < 8; ++c) {
        v8h x0 = *(const v8h*)(qr + c * 32 + ah);
        v8h x1 = *(const v8h*)(qr + c * 32 + 8 + ah);
        for (int t = 0; t < 8; ++t) { aq[c][t] = x0[t]; aq[c][8 + t] = x1[t]; }
    }

    v8f oacc[16];
    for (int t = 0; t < 16; ++t) oacc[t] = (v8f){0.f,0.f,0.f,0.f,0.f,0.f,0.f,0.f};
    float mi[8], li[8];
    for (int r = 0; r < 8; ++r) { mi[r] = -1e30f; li[r] = 0.f; }

    for (int j = 0; j < NN / 32; ++j) {
        const int k0 = j * 32;
        // ---- prefetch next tile into cache while we stage/compute this one ----
        if (j + 1 < NN / 32) {
            __builtin_prefetch(kb + (size_t)(k0 + 32) * DQ + tid * 32, 0, 1); // 256*64B = next K tile
            __builtin_prefetch(vb + (size_t)tid * NN + (k0 + 32), 0, 1);      // next V tile row segs
        }
        // ---- cooperative staging: K tile is one contiguous 16KB block ----
        {
            const _Float16* src = kb + (size_t)k0 * DQ;
            for (int t = 0; t < 4; ++t) {
                const int idx = tid + t * 256;   // 16B chunk id, 0..1023
                *(float4*)&kv[KOFF + idx * 8] = *(const float4*)(src + idx * 8);
            }
            for (int t = 0; t < 4; ++t) {
                const int idx = tid + t * 256;
                const int d = idx >> 2, c = idx & 3;  // 256 d-rows x 4 chunks
                *(float4*)&kv[VOFF + d * 32 + c * 8] =
                    *(const float4*)(vb + (size_t)d * NN + k0 + c * 8);
            }
        }
        __syncthreads();
        // ---- scores: two 16x16 tiles covering 32 keys, K-dim = 256 ----
        v8f s0 = {0.f,0.f,0.f,0.f,0.f,0.f,0.f,0.f};
        v8f s1 = {0.f,0.f,0.f,0.f,0.f,0.f,0.f,0.f};
        const _Float16* kr0 = &kv[KOFF + lm * DQ];
        const _Float16* kr1 = &kv[KOFF + (16 + lm) * DQ];
        for (int c = 0; c < 8; ++c) {
            v16h bk0 = *(const v16h*)(kr0 + c * 32 + bh);
            v16h bk1 = *(const v16h*)(kr1 + c * 32 + bh);
            s0 = wmma_f16(aq[c], bk0, s0);
            s1 = wmma_f16(aq[c], bk1, s1);
        }
        // ---- online softmax (row stats live in C-matrix lane layout) ----
        float alpha[8];
        for (int r = 0; r < 8; ++r) {
            float tm = fmaxf(s0[r], s1[r]);
            tm = fmaxf(tm, __shfl_xor(tm, 1, 32));
            tm = fmaxf(tm, __shfl_xor(tm, 2, 32));
            tm = fmaxf(tm, __shfl_xor(tm, 4, 32));
            tm = fmaxf(tm, __shfl_xor(tm, 8, 32));
            const float mn = fmaxf(mi[r], tm);
            const float al = __expf(mi[r] - mn);
            const float p0 = __expf(s0[r] - mn);
            const float p1 = __expf(s1[r] - mn);
            s0[r] = p0; s1[r] = p1;
            float rs = p0 + p1;
            rs += __shfl_xor(rs, 1, 32);
            rs += __shfl_xor(rs, 2, 32);
            rs += __shfl_xor(rs, 4, 32);
            rs += __shfl_xor(rs, 8, 32);
            li[r] = li[r] * al + rs;
            mi[r] = mn;
            alpha[r] = al;
        }
        for (int t = 0; t < 16; ++t)
            for (int r = 0; r < 8; ++r) oacc[t][r] *= alpha[r];
        // ---- transpose P (C-layout -> A-layout) through private LDS buf ----
        for (int r = 0; r < 8; ++r) {
            pbuf[(r + mo) * 32 + lm]      = (_Float16)s0[r];
            pbuf[(r + mo) * 32 + 16 + lm] = (_Float16)s1[r];
        }
        v16h ap;
        {
            const _Float16* pr = pbuf + lm * 32;
            v8h x0 = *(const v8h*)(pr + ah);
            v8h x1 = *(const v8h*)(pr + 8 + ah);
            for (int t = 0; t < 8; ++t) { ap[t] = x0[t]; ap[8 + t] = x1[t]; }
        }
        // ---- O += P @ V from the shared V tile ----
        for (int t = 0; t < 16; ++t) {
            v16h bv_ = *(const v16h*)&kv[VOFF + (t * 16 + lm) * 32 + bh];
            oacc[t] = wmma_f16(ap, bv_, oacc[t]);
        }
        __syncthreads();   // protect K/V tile before next staging pass
    }

    // ---- normalize and build A-operand chunks of O via private LDS buf ----
    float inv[8];
    for (int r = 0; r < 8; ++r) inv[r] = 1.f / li[r];
    v16h ao[8];
    for (int c = 0; c < 8; ++c) {
        for (int r = 0; r < 8; ++r) {
            pbuf[(r + mo) * 32 + lm]      = (_Float16)(oacc[2 * c][r]     * inv[r]);
            pbuf[(r + mo) * 32 + 16 + lm] = (_Float16)(oacc[2 * c + 1][r] * inv[r]);
        }
        const _Float16* pr = pbuf + lm * 32;
        v8h x0 = *(const v8h*)(pr + ah);
        v8h x1 = *(const v8h*)(pr + 8 + ah);
        for (int t = 0; t < 8; ++t) { ao[c][t] = x0[t]; ao[c][8 + t] = x1[t]; }
    }
    // ---- fused output projection: [16,256] @ net_W^T -> [16,128] ----
    for (int t2 = 0; t2 < 8; ++t2) {
        const int n0 = t2 * 16;
        v8f acc = {0.f,0.f,0.f,0.f,0.f,0.f,0.f,0.f};
        const _Float16* wr = Wnh + (size_t)(n0 + lm) * DQ;
        for (int c = 0; c < 8; ++c) {
            v16h bw = *(const v16h*)(wr + c * 32 + bh);
            acc = wmma_f16(ao[c], bw, acc);
        }
        const float bb = net_b[n0 + lm];
        for (int r = 0; r < 8; ++r)
            out[((size_t)b * NROWS + q0 + r + mo) * HIDN + n0 + lm] = acc[r] + bb;
    }
}

// ---------------------------------------------------------------------------
// Tiny sfc transformer branch: one block per batch, plain f32 in LDS.
// ---------------------------------------------------------------------------
__global__ void sfc_kernel(const float* __restrict__ sfc_state,
                           const int* __restrict__ node_pair,
                           const float* __restrict__ node_embed,
                           const float* __restrict__ sfc_W, const float* __restrict__ sfc_b,
                           const float* __restrict__ pos_embed,
                           const float* __restrict__ qkv_w, const float* __restrict__ qkv_b,
                           const float* __restrict__ out_w, const float* __restrict__ out_b,
                           const float* __restrict__ ln1_w, const float* __restrict__ ln1_b,
                           const float* __restrict__ lin1_w, const float* __restrict__ lin1_b,
                           const float* __restrict__ lin2_w, const float* __restrict__ lin2_b,
                           const float* __restrict__ ln2_w, const float* __restrict__ ln2_b,
                           float* __restrict__ out) {
    const int b = blockIdx.x;
    const int tid = threadIdx.x;             // 128 threads
    __shared__ float x[SEQL][HIDN];
    __shared__ float sfc[SEQL][32];
    __shared__ float qkv[SEQL][3 * HIDN];
    __shared__ float attn[SEQL][HIDN];
    __shared__ float scores[4][SEQL][SEQL];
    __shared__ float pad[SEQL];

    for (int i = tid; i < SEQL * 32; i += blockDim.x) {
        const int s = i / 32, d = i % 32;
        float v;
        if (s == 0)             v = node_embed[node_pair[b * 2 + 0] * 32 + d];
        else if (s == SEQL - 1) v = node_embed[node_pair[b * 2 + 1] * 32 + d];
        else                    v = sfc_state[((size_t)b * 16 + (s - 1)) * 32 + d];
        sfc[s][d] = v;
    }
    __syncthreads();
    if (tid < SEQL) {
        float sa = 0.f;
        for (int d = 0; d < 32; ++d) sa += fabsf(sfc[tid][d]);
        pad[tid] = (sa == 0.f) ? -1e9f : 0.f;
    }
    __syncthreads();
    for (int i = tid; i < SEQL * HIDN; i += blockDim.x) {
        const int s = i / HIDN, h = i % HIDN;
        float acc = sfc_b[h];
        for (int d = 0; d < 32; ++d) acc += sfc[s][d] * sfc_W[h * 32 + d];
        x[s][h] = acc + pos_embed[s * HIDN + h];
    }
    __syncthreads();

    for (int l = 0; l < 2; ++l) {
        const float* qw = qkv_w + (size_t)l * 3 * HIDN * HIDN;
        const float* qb = qkv_b + (size_t)l * 3 * HIDN;
        for (int i = tid; i < SEQL * 3 * HIDN; i += blockDim.x) {
            const int s = i / (3 * HIDN), j = i % (3 * HIDN);
            float acc = qb[j];
            for (int h = 0; h < HIDN; ++h) acc += x[s][h] * qw[j * HIDN + h];
            qkv[s][j] = acc;
        }
        __syncthreads();
        for (int i = tid; i < 4 * SEQL * SEQL; i += blockDim.x) {
            const int h = i / (SEQL * SEQL), q = (i / SEQL) % SEQL, k = i % SEQL;
            const float* qp = &qkv[q][h * 32];
            const float* kp = &qkv[k][HIDN + h * 32];
            float acc = 0.f;
            for (int d = 0; d < 32; ++d) acc += qp[d] * kp[d];
            scores[h][q][k] = acc * 0.17677669529663689f + pad[k];
        }
        __syncthreads();
        for (int i = tid; i < 4 * SEQL; i += blockDim.x) {
            const int h = i / SEQL, q = i % SEQL;
            float mx = -1e30f;
            for (int k = 0; k < SEQL; ++k) mx = fmaxf(mx, scores[h][q][k]);
            float sm = 0.f;
            for (int k = 0; k < SEQL; ++k) { float e = __expf(scores[h][q][k] - mx); scores[h][q][k] = e; sm += e; }
            const float inv = 1.f / sm;
            for (int k = 0; k < SEQL; ++k) scores[h][q][k] *= inv;
        }
        __syncthreads();
        for (int i = tid; i < SEQL * HIDN; i += blockDim.x) {
            const int s = i / HIDN, c = i % HIDN, h = c / 32, d = c % 32;
            float acc = 0.f;
            for (int k = 0; k < SEQL; ++k) acc += scores[h][s][k] * qkv[k][2 * HIDN + h * 32 + d];
            attn[s][c] = acc;
        }
        __syncthreads();
        const float* ow = out_w + (size_t)l * HIDN * HIDN;
        const float* ob = out_b + (size_t)l * HIDN;
        for (int i = tid; i < SEQL * HIDN; i += blockDim.x) {
            const int s = i / HIDN, h = i % HIDN;
            float acc = ob[h];
            for (int j = 0; j < HIDN; ++j) acc += attn[s][j] * ow[h * HIDN + j];
            qkv[s][h] = x[s][h] + acc;   // residual temp
        }
        __syncthreads();
        if (tid < SEQL) {
            const int s = tid;
            float mu = 0.f;
            for (int h = 0; h < HIDN; ++h) mu += qkv[s][h];
            mu /= HIDN;
            float var = 0.f;
            for (int h = 0; h < HIDN; ++h) { float d = qkv[s][h] - mu; var += d * d; }
            var /= HIDN;
            const float rs = rsqrtf(var + 1e-5f);
            for (int h = 0; h < HIDN; ++h)
                x[s][h] = (qkv[s][h] - mu) * rs * ln1_w[l * HIDN + h] + ln1_b[l * HIDN + h];
        }
        __syncthreads();
        const float* w1 = lin1_w + (size_t)l * 8 * HIDN;
        const float* b1 = lin1_b + (size_t)l * 8;
        const float* w2 = lin2_w + (size_t)l * HIDN * 8;
        const float* b2 = lin2_b + (size_t)l * HIDN;
        for (int i = tid; i < SEQL * 8; i += blockDim.x) {
            const int s = i / 8, f = i % 8;
            float acc = b1[f];
            for (int h = 0; h < HIDN; ++h) acc += x[s][h] * w1[f * HIDN + h];
            attn[s][f] = fmaxf(acc, 0.f);
        }
        __syncthreads();
        for (int i = tid; i < SEQL * HIDN; i += blockDim.x) {
            const int s = i / HIDN, h = i % HIDN;
            float acc = b2[h];
            for (int f = 0; f < 8; ++f) acc += attn[s][f] * w2[h * 8 + f];
            qkv[s][h] = x[s][h] + acc;
        }
        __syncthreads();
        if (tid < SEQL) {
            const int s = tid;
            float mu = 0.f;
            for (int h = 0; h < HIDN; ++h) mu += qkv[s][h];
            mu /= HIDN;
            float var = 0.f;
            for (int h = 0; h < HIDN; ++h) { float d = qkv[s][h] - mu; var += d * d; }
            var /= HIDN;
            const float rs = rsqrtf(var + 1e-5f);
            for (int h = 0; h < HIDN; ++h)
                x[s][h] = (qkv[s][h] - mu) * rs * ln2_w[l * HIDN + h] + ln2_b[l * HIDN + h];
        }
        __syncthreads();
    }
    for (int i = tid; i < SEQL * HIDN; i += blockDim.x) {
        const int s = i / HIDN, h = i % HIDN;
        out[((size_t)b * NROWS + NN + s) * HIDN + h] = x[s][h];
    }
}

// ---------------------------------------------------------------------------
extern "C" void kernel_launch(void* const* d_in, const int* in_sizes, int n_in,
                              void* d_out, int out_size, void* d_ws, size_t ws_size,
                              hipStream_t stream) {
    const float* net_state  = (const float*)d_in[0];
    const float* sfc_state  = (const float*)d_in[1];
    const int*   node_pair  = (const int*)d_in[2];
    const float* Wq = (const float*)d_in[3];
    const float* bq = (const float*)d_in[4];
    const float* Wk = (const float*)d_in[5];
    const float* bk = (const float*)d_in[6];
    const float* Wv = (const float*)d_in[7];
    const float* bv = (const float*)d_in[8];
    const float* net_W = (const float*)d_in[9];
    const float* net_b = (const float*)d_in[10];
    const float* node_embed = (const float*)d_in[11];
    const float* sfc_W = (const float*)d_in[12];
    const float* sfc_b = (const float*)d_in[13];
    const float* pos_embed = (const float*)d_in[14];
    const float* qkv_w = (const float*)d_in[15];
    const float* qkv_b = (const float*)d_in[16];
    const float* out_w = (const float*)d_in[17];
    const float* out_b = (const float*)d_in[18];
    const float* ln1_w = (const float*)d_in[19];
    const float* ln1_b = (const float*)d_in[20];
    const float* lin1_w = (const float*)d_in[21];
    const float* lin1_b = (const float*)d_in[22];
    const float* lin2_w = (const float*)d_in[23];
    const float* lin2_b = (const float*)d_in[24];
    const float* ln2_w = (const float*)d_in[25];
    const float* ln2_b = (const float*)d_in[26];
    float* out = (float*)d_out;

    char* ws = (char*)d_ws;
    _Float16* Qh  = (_Float16*)(ws);
    _Float16* Kh  = (_Float16*)(ws + (size_t)16 * 1024 * 1024);
    _Float16* Vth = (_Float16*)(ws + (size_t)32 * 1024 * 1024);
    _Float16* Wqh = (_Float16*)(ws + (size_t)48 * 1024 * 1024);
    _Float16* Wkh = Wqh + DQ * DN;
    _Float16* Wvh = Wkh + DQ * DN;
    _Float16* Wnh = Wvh + DQ * DN;

    cvt_weights<<<128, 256, 0, stream>>>(Wq, Wk, Wv, net_W, Wqh, Wkh, Wvh, Wnh);
    qkv_proj<<<BATCH * NN / 16, 32, 0, stream>>>(net_state, Wqh, Wkh, Wvh,
                                                 bq, bk, bv, Qh, Kh, Vth);
    flash_kernel<<<BATCH * NN / QROWS, WAVES * 32, 0, stream>>>(Qh, Kh, Vth, Wnh,
                                                                net_b, out);
    sfc_kernel<<<BATCH, 128, 0, stream>>>(sfc_state, node_pair, node_embed,
                                          sfc_W, sfc_b, pos_embed,
                                          qkv_w, qkv_b, out_w, out_b,
                                          ln1_w, ln1_b, lin1_w, lin1_b,
                                          lin2_w, lin2_b, ln2_w, ln2_b, out);
}
